// AttentionHead_42004780155494
// MI455X (gfx1250) — matrix operand order repaired and checked by reading, not compiled, not run on hardware
//
#include <hip/hip_runtime.h>
#include <hip/hip_bf16.h>
#include <math.h>

// Problem constants (reference: B,T,C,H = 8,2048,1024,128)
#define Bb 8
#define Tt 2048
#define Cc 1024
#define Hh 128
#define LDP 2064   // LDS row stride (floats): 2048 + 16 pad (bank-conflict avoidance)

typedef __attribute__((ext_vector_type(16))) __bf16 v16bf;
typedef __attribute__((ext_vector_type(4)))  __bf16 v4bf;
typedef __attribute__((ext_vector_type(8)))  float  v8f;

__device__ __forceinline__ v8f wmma_bf16(v16bf a, v16bf b, v8f c) {
  // v_wmma_f32_16x16x32_bf16: (neg_a, A, neg_b, B, c_mod, C, reuse_a, reuse_b)
  return __builtin_amdgcn_wmma_f32_16x16x32_bf16(false, a, false, b, (short)0, c, false, false);
}

// A-fragment (16x32 bf16, M x K), bf16 source, row-major contiguous K.
// lanes 0-15: row M=lane, halves 0..7 = K k0..k0+7, halves 8..15 = K k0+16..k0+23
// lanes 16-31: row M=lane-16, K ranges shifted by +8.
__device__ __forceinline__ v16bf fragA_bf(const __bf16* __restrict__ p, int ld,
                                          int row, int k0, int lane) {
  const __bf16* q = p + (size_t)row * ld + k0 + ((lane & 16) ? 8 : 0);
  v16bf a;
#pragma unroll
  for (int i = 0; i < 8; ++i) a[i] = q[i];
#pragma unroll
  for (int i = 0; i < 8; ++i) a[8 + i] = q[16 + i];
  return a;
}

// Same fragment from an f32 source (fallback path only), cast to bf16.
__device__ __forceinline__ v16bf fragA_f32(const float* __restrict__ p, int ld,
                                           int row, int k0, int lane) {
  const float* q = p + (size_t)row * ld + k0 + ((lane & 16) ? 8 : 0);
  v16bf a;
#pragma unroll
  for (int i = 0; i < 8; ++i) a[i] = (__bf16)q[i];
#pragma unroll
  for (int i = 0; i < 8; ++i) a[8 + i] = (__bf16)q[16 + i];
  return a;
}

// B-fragment (32x16 bf16, K x N), source stored N-major (row = N, contiguous K):
// lane holds column N=lane&15; lanes 0-15 K=k0..k0+7,k0+16..k0+23; lanes 16-31 shifted by +8.
__device__ __forceinline__ v16bf fragB_bf(const __bf16* __restrict__ p, int ld,
                                          int n, int k0, int lane) {
  const __bf16* q = p + (size_t)n * ld + k0 + ((lane & 16) ? 8 : 0);
  v16bf a;
#pragma unroll
  for (int i = 0; i < 8; ++i) a[i] = q[i];
#pragma unroll
  for (int i = 0; i < 8; ++i) a[8 + i] = q[16 + i];
  return a;
}

// ---------------------------------------------------------------------------
// Kernel 0: streaming f32 -> bf16 conversion (x and weights), vectorized x4.
// ---------------------------------------------------------------------------
__global__ __launch_bounds__(256) void cvt_kernel(const float* __restrict__ in,
                                                  __bf16* __restrict__ out, int n) {
  const int i = (blockIdx.x * 256 + threadIdx.x) * 4;
  if (i < n) {
    const float4 v = *(const float4*)(in + i);
    v4bf o;
    o[0] = (__bf16)v.x; o[1] = (__bf16)v.y; o[2] = (__bf16)v.z; o[3] = (__bf16)v.w;
    *(v4bf*)(out + i) = o;
  }
}

// ---------------------------------------------------------------------------
// Kernel 1: fused QKV projection, all-bf16 operands. grid.x = (B*T)/16,
// block = 256 (8 waves); wave w computes N-tile [16w,16w+16) of H for Q,K,V.
// Adds positional term to Q,K; stores Q,K bf16 (B*T,H) and V transposed (B,H,T).
// ---------------------------------------------------------------------------
__global__ __launch_bounds__(256) void qkv_kernel(
    const __bf16* __restrict__ xb, const __bf16* __restrict__ wq,
    const __bf16* __restrict__ wk, const __bf16* __restrict__ wv,
    __bf16* __restrict__ qb, __bf16* __restrict__ kb, __bf16* __restrict__ vT) {
  const int lane = threadIdx.x & 31;
  const int wave = threadIdx.x >> 5;
  const int m0 = blockIdx.x * 16;
  const int n0 = wave * 16;
  const int arow = m0 + (lane & 15);
  const int brow = n0 + (lane & 15);

  v8f aq = {0.f,0.f,0.f,0.f,0.f,0.f,0.f,0.f};
  v8f ak = aq, av = aq;

  for (int k0 = 0; k0 < Cc; k0 += 32) {
    v16bf a  = fragA_bf(xb, Cc, arow, k0, lane);
    v16bf bq = fragB_bf(wq, Cc, brow, k0, lane);
    aq = wmma_bf16(a, bq, aq);
    v16bf bk = fragB_bf(wk, Cc, brow, k0, lane);
    ak = wmma_bf16(a, bk, ak);
    v16bf bv = fragB_bf(wv, Cc, brow, k0, lane);
    av = wmma_bf16(a, bv, av);
  }

  const int n = n0 + (lane & 15);
#pragma unroll
  for (int j = 0; j < 8; ++j) {
    const int m = m0 + j + ((lane & 16) ? 8 : 0);   // C/D layout: VGPR j -> M=j / j+8
    const int t = m & (Tt - 1);
    const int b = m >> 11;                          // T = 2048 = 2^11
    const float pos = (float)t;
    qb[(size_t)m * Hh + n] = (__bf16)(aq[j] + pos);
    kb[(size_t)m * Hh + n] = (__bf16)(ak[j] + pos);
    vT[((size_t)b * Hh + n) * Tt + t] = (__bf16)av[j];
  }
}

// ---------------------------------------------------------------------------
// Kernel 2: fused scores + causal softmax. grid = (T/16, B); block = 256.
// Score tile S (16 x 2048 f32) lives in LDS (132 KB; gfx1250 WGP has 320 KB).
// Phase 1: waves cooperatively compute S = (Q+pos)(K+pos)^T / 32 over the
//          causal col-tiles only. Phase 2: per-row softmax; writes normalized
//          probs as f32 (output) and bf16 (workspace, for the attn@V GEMM).
// ---------------------------------------------------------------------------
__global__ __launch_bounds__(256) void scores_softmax_kernel(
    const __bf16* __restrict__ qb, const __bf16* __restrict__ kb,
    float* __restrict__ attn, __bf16* __restrict__ attnbf, int use_bf) {
  __shared__ float S[16 * LDP];                    // 132096 bytes
  const int lane = threadIdx.x & 31;
  const int wave = threadIdx.x >> 5;
  const int m0 = blockIdx.x * 16;
  const int b  = blockIdx.y;
  const __bf16* q = qb + (size_t)b * Tt * Hh;
  const __bf16* k = kb + (size_t)b * Tt * Hh;

  v16bf a[4];
#pragma unroll
  for (int kc = 0; kc < 4; ++kc)
    a[kc] = fragA_bf(q, Hh, m0 + (lane & 15), kc * 32, lane);

  const int nst = blockIdx.x + 1;                  // causal: col tiles with s0 <= m0+15
  for (int st = wave; st < nst; st += 8) {
    const int s0 = st * 16;
    v8f acc = {0.f,0.f,0.f,0.f,0.f,0.f,0.f,0.f};
#pragma unroll
    for (int kc = 0; kc < 4; ++kc) {
      v16bf bk = fragB_bf(k, Hh, s0 + (lane & 15), kc * 32, lane);
      acc = wmma_bf16(a[kc], bk, acc);
    }
#pragma unroll
    for (int j = 0; j < 8; ++j) {
      const int mrel = j + ((lane & 16) ? 8 : 0);
      S[mrel * LDP + s0 + (lane & 15)] = acc[j] * 0.03125f;   // 1/sqrt(1024)
    }
  }
  __syncthreads();

  // Phase 2: each wave owns 2 rows (wave-private; no further barriers needed).
#pragma unroll
  for (int rr = 0; rr < 2; ++rr) {
    const int r = wave * 2 + rr;
    const int t = m0 + r;
    float* Sr = &S[r * LDP];

    float mx = -INFINITY;
    for (int i = lane; i <= t; i += 32) mx = fmaxf(mx, Sr[i]);
#pragma unroll
    for (int off = 16; off; off >>= 1) mx = fmaxf(mx, __shfl_xor(mx, off));

    float sum = 0.f;
    for (int i = lane; i <= t; i += 32) {
      const float e = __expf(Sr[i] - mx);
      Sr[i] = e;
      sum += e;
    }
#pragma unroll
    for (int off = 16; off; off >>= 1) sum += __shfl_xor(sum, off);
    const float inv = 1.0f / sum;

    float*  po = attn   + ((size_t)b * Tt + t) * Tt;
    __bf16* pb = attnbf + ((size_t)b * Tt + t) * Tt;
    for (int i = lane; i < Tt; i += 32) {
      const float v = (i <= t) ? Sr[i] * inv : 0.f;
      po[i] = v;
      if (use_bf) pb[i] = (__bf16)v;
    }
  }
}

// ---------------------------------------------------------------------------
// Kernel 3: res = attn @ V. grid = (T/16, B); block = 256; wave w covers the
// 16-wide H tile [16w,16w+16). k-loop truncated at the causal boundary.
// Primary variant reads the bf16 prob copy; fallback reads f32 output probs.
// ---------------------------------------------------------------------------
__global__ __launch_bounds__(256) void out_kernel_bf(
    const __bf16* __restrict__ attnbf, const __bf16* __restrict__ vT,
    float* __restrict__ res) {
  const int lane = threadIdx.x & 31;
  const int wave = threadIdx.x >> 5;
  const int m0 = blockIdx.x * 16;
  const int b  = blockIdx.y;
  const int n0 = wave * 16;

  const __bf16* A  = attnbf + ((size_t)b * Tt + m0) * Tt;
  const __bf16* Vt = vT + (size_t)b * Hh * Tt;

  v8f acc = {0.f,0.f,0.f,0.f,0.f,0.f,0.f,0.f};
  const int nk = (m0 + 47) >> 5;                  // 32-chunks up to causal boundary
  const int arow = lane & 15;
  const int brow = n0 + (lane & 15);
  for (int kc = 0; kc < nk; ++kc) {
    v16bf a  = fragA_bf(A,  Tt, arow, kc * 32, lane);
    v16bf bv = fragB_bf(Vt, Tt, brow, kc * 32, lane);
    acc = wmma_bf16(a, bv, acc);
  }

  float* out = res + ((size_t)b * Tt + m0) * Hh;
#pragma unroll
  for (int j = 0; j < 8; ++j) {
    const int mrel = j + ((lane & 16) ? 8 : 0);
    out[(size_t)mrel * Hh + n0 + (lane & 15)] = acc[j];
  }
}

__global__ __launch_bounds__(256) void out_kernel_f32(
    const float* __restrict__ attn, const __bf16* __restrict__ vT,
    float* __restrict__ res) {
  const int lane = threadIdx.x & 31;
  const int wave = threadIdx.x >> 5;
  const int m0 = blockIdx.x * 16;
  const int b  = blockIdx.y;
  const int n0 = wave * 16;

  const float*  A  = attn + ((size_t)b * Tt + m0) * Tt;
  const __bf16* Vt = vT + (size_t)b * Hh * Tt;

  v8f acc = {0.f,0.f,0.f,0.f,0.f,0.f,0.f,0.f};
  const int nk = (m0 + 47) >> 5;
  const int arow = lane & 15;
  const int brow = n0 + (lane & 15);
  for (int kc = 0; kc < nk; ++kc) {
    v16bf a  = fragA_f32(A,  Tt, arow, kc * 32, lane);
    v16bf bv = fragB_bf(Vt, Tt, brow, kc * 32, lane);
    acc = wmma_bf16(a, bv, acc);
  }

  float* out = res + ((size_t)b * Tt + m0) * Hh;
#pragma unroll
  for (int j = 0; j < 8; ++j) {
    const int mrel = j + ((lane & 16) ? 8 : 0);
    out[(size_t)mrel * Hh + n0 + (lane & 15)] = acc[j];
  }
}

// ---------------------------------------------------------------------------
extern "C" void kernel_launch(void* const* d_in, const int* in_sizes, int n_in,
                              void* d_out, int out_size, void* d_ws, size_t ws_size,
                              hipStream_t stream) {
  const float* x  = (const float*)d_in[0];
  const float* Wq = (const float*)d_in[1];
  const float* Wk = (const float*)d_in[2];
  const float* Wv = (const float*)d_in[3];

  float* res  = (float*)d_out;                       // (B,T,H)
  float* attn = res + (size_t)Bb * Tt * Hh;          // (B,T,T)

  // Workspace layout (bytes):
  char* ws = (char*)d_ws;
  const size_t szQ  = (size_t)Bb * Tt * Hh * 2;      // 4 MB each
  const size_t szW  = (size_t)Hh * Cc * 2;           // 256 KB each
  const size_t szX  = (size_t)Bb * Tt * Cc * 2;      // 33.5 MB
  const size_t szAB = (size_t)Bb * Tt * Tt * 2;      // 67 MB

  __bf16* qb  = (__bf16*)(ws);
  __bf16* kb  = (__bf16*)(ws + szQ);
  __bf16* vT  = (__bf16*)(ws + 2 * szQ);
  __bf16* wqb = (__bf16*)(ws + 3 * szQ);
  __bf16* wkb = (__bf16*)(ws + 3 * szQ + szW);
  __bf16* wvb = (__bf16*)(ws + 3 * szQ + 2 * szW);
  __bf16* xb  = (__bf16*)(ws + 3 * szQ + 3 * szW);
  __bf16* attnbf = (__bf16*)(ws + 3 * szQ + 3 * szW + szX);

  const int use_bf = (ws_size >= 3 * szQ + 3 * szW + szX + szAB) ? 1 : 0;

  const int nx = Bb * Tt * Cc;                       // 16,777,216
  const int nw = Hh * Cc;                            // 131,072
  cvt_kernel<<<nx / (256 * 4), 256, 0, stream>>>(x, xb, nx);
  cvt_kernel<<<nw / (256 * 4), 256, 0, stream>>>(Wq, wqb, nw);
  cvt_kernel<<<nw / (256 * 4), 256, 0, stream>>>(Wk, wkb, nw);
  cvt_kernel<<<nw / (256 * 4), 256, 0, stream>>>(Wv, wvb, nw);

  qkv_kernel<<<(Bb * Tt) / 16, 256, 0, stream>>>(xb, wqb, wkb, wvb, qb, kb, vT);

  scores_softmax_kernel<<<dim3(Tt / 16, Bb), 256, 0, stream>>>(
      qb, kb, attn, use_bf ? attnbf : (__bf16*)attn, use_bf);

  if (use_bf)
    out_kernel_bf<<<dim3(Tt / 16, Bb), 256, 0, stream>>>(attnbf, vT, res);
  else
    out_kernel_f32<<<dim3(Tt / 16, Bb), 256, 0, stream>>>(attn, vT, res);
}
